// SpatiotemporalGraphTransformer_91207925498460
// MI455X (gfx1250) — matrix operand order repaired and checked by reading, not compiled, not run on hardware
//
#include <hip/hip_runtime.h>

// ---------------------------------------------------------------------------
// MI455X (gfx1250) implementation of the spatiotemporal graph transformer.
// Dense GEMMs: LDS-staged, double-buffered tiles moved with gfx1250 async
// global->LDS copies (ASYNCcnt), fragments fed from LDS into
// v_wmma_f32_16x16x32_f16 (f16 in, f32 accum). Graph-irregular ops are
// wave32 VALU kernels with float atomics.
// ---------------------------------------------------------------------------

typedef _Float16 h8v  __attribute__((ext_vector_type(8)));
typedef _Float16 h16v __attribute__((ext_vector_type(16)));
typedef float    f8v  __attribute__((ext_vector_type(8)));
typedef int      v4i  __attribute__((vector_size(16)));      // matches builtin param type
typedef v4i __attribute__((address_space(1))) v4i_g;         // global (AS1)
typedef v4i __attribute__((address_space(3))) v4i_l;         // LDS (AS3)

union HFrag { h16v v; h8v h[2]; };

// ---- gfx1250 async global->LDS copy (ASYNCcnt) with sync fallback --------
#if __has_builtin(__builtin_amdgcn_global_load_async_to_lds_b128)
#define ASYNC_COPY_AVAILABLE 1
__device__ __forceinline__ void async_copy16(const _Float16* g, _Float16* l)
{
    __builtin_amdgcn_global_load_async_to_lds_b128(
        (v4i_g*)(uintptr_t)g,
        (v4i_l*)(unsigned int)(uintptr_t)l,
        0, 0);
}
#else
__device__ __forceinline__ void async_copy16(const _Float16* g, _Float16* l)
{
    *(h8v*)l = *(const h8v*)g;   // synchronous fallback: global load + ds store
}
#endif

#if __has_builtin(__builtin_amdgcn_s_wait_asynccnt)
#define WAIT_ASYNC0() __builtin_amdgcn_s_wait_asynccnt(0)
#define WAIT_ASYNC3() __builtin_amdgcn_s_wait_asynccnt(3)
#elif defined(ASYNC_COPY_AVAILABLE)
#define WAIT_ASYNC0() asm volatile("s_wait_asynccnt 0x0" ::: "memory")
#define WAIT_ASYNC3() asm volatile("s_wait_asynccnt 0x3" ::: "memory")
#else
#define WAIT_ASYNC0() ((void)0)
#define WAIT_ASYNC3() ((void)0)
#endif

// ---------------------------------------------------------------------------
// Generic batched WMMA GEMM:  C[z] = act(A[z] (MxK) * BT[z]^T (KxN) + bias)
//   A  : f16 row-major, leading dim lda   (lda % 8 == 0)
//   BT : f16 [N,K] row-major (i.e. W^T),  leading dim ldb (ldb % 8 == 0)
//   C  : f32 row-major, leading dim ldc
//   z = zo*innerN + zi ; per-matrix offsets = zo*s?o + zi*s?i (elements)
//   M % 128 == 0 ; N % 16 == 0 ; K % 32 == 0
// Block: 256 thr = 8 waves as 4(M)x2(N) -> block tile 128M x 64N.
// Per k-step(32): A tile 128x32 + B tile 64x32 staged in LDS (double
// buffered, async copies), each wave computes a 32x32 tile = 2x2 WMMA.
// ---------------------------------------------------------------------------
__global__ __launch_bounds__(256) void gemm_wmma_kernel(
    const _Float16* __restrict__ A, int lda, long sAo, long sAi,
    const _Float16* __restrict__ BT, int ldb, long sBo, long sBi,
    float* __restrict__ C, int ldc, long sCo, long sCi,
    const float* __restrict__ bias,
    int M, int N, int K, int act, int innerN)
{
    __shared__ _Float16 Asb[2][128 * 32];
    __shared__ _Float16 Bsb[2][64 * 32];

    int z  = blockIdx.z;
    int zo = z / innerN;
    int zi = z - zo * innerN;
    A  += zo * sAo + zi * sAi;
    BT += zo * sBo + zi * sBi;
    C  += zo * sCo + zi * sCi;

    const int tid  = threadIdx.x;
    const int wave = __builtin_amdgcn_readfirstlane(tid >> 5);   // wave-uniform SGPR
    const int lane = tid & 31;
    const int mBase = (wave >> 1) * 32;            // within 128-row block tile
    const int nBase = (wave & 1) * 32;             // within 64-col block tile
    const int mBlock = blockIdx.x * 128;
    const int nBlock = blockIdx.y * 64;
    const int lm   = lane & 15;
    const int hi   = lane >> 4;
    const int lk8  = hi * 8;                       // A fragment K sub-offset
    const int lk16 = hi * 16;                      // unused: B fragments use lk8 like A

    // scalar (SALU) tile-validity guards -> no EXEC manipulation around WMMA
    const bool n0 = (nBlock + nBase      < N);
    const bool n1 = (nBlock + nBase + 16 < N);

    // stage one 32-wide K-slice of A(128 rows) + B(64 rows) into LDS buffer.
    // 192 rows * 4 chunks(16B) = 768 chunks -> 3 per thread.
    auto stage = [&](int buf, int k0) {
        #pragma unroll
        for (int i = 0; i < 3; ++i) {
            int c   = tid + i * 256;
            int row = c >> 2;
            int sub = (c & 3) * 8;
            if (row < 128) {
                async_copy16(A + (long)(mBlock + row) * lda + k0 + sub,
                             &Asb[buf][row * 32 + sub]);
            } else {
                int br = row - 128;
                int gn = nBlock + br;
                if (gn > N - 1) gn = N - 1;        // clamp (stay in-bounds)
                async_copy16(BT + (long)gn * ldb + k0 + sub,
                             &Bsb[buf][br * 32 + sub]);
            }
        }
    };

    f8v acc00{}, acc01{}, acc10{}, acc11{};
    const int ntiles = K >> 5;

    stage(0, 0);
    for (int t = 0; t < ntiles; ++t) {
        const int buf = t & 1;
        if (t + 1 < ntiles) {
            stage(buf ^ 1, (t + 1) * 32);          // prefetch next K-slice
            WAIT_ASYNC3();                          // drain only the older batch
        } else {
            WAIT_ASYNC0();
        }
        __syncthreads();

        const _Float16* Ab = &Asb[buf][0];
        const _Float16* Bb = &Bsb[buf][0];
        HFrag a0, a1, b0, b1;
        a0.h[0] = *(const h8v*)(Ab + (mBase      + lm) * 32 + lk8);
        a0.h[1] = *(const h8v*)(Ab + (mBase      + lm) * 32 + 16 + lk8);
        a1.h[0] = *(const h8v*)(Ab + (mBase + 16 + lm) * 32 + lk8);
        a1.h[1] = *(const h8v*)(Ab + (mBase + 16 + lm) * 32 + 16 + lk8);
        b0.h[0] = *(const h8v*)(Bb + (nBase      + lm) * 32 + lk8);
        b0.h[1] = *(const h8v*)(Bb + (nBase      + lm) * 32 + 16 + lk8);
        b1.h[0] = *(const h8v*)(Bb + (nBase + 16 + lm) * 32 + lk8);
        b1.h[1] = *(const h8v*)(Bb + (nBase + 16 + lm) * 32 + 16 + lk8);

        if (n0) {
            acc00 = __builtin_amdgcn_wmma_f32_16x16x32_f16(false, a0.v, false, b0.v,
                                                           (short)0, acc00, false, false);
            acc10 = __builtin_amdgcn_wmma_f32_16x16x32_f16(false, a1.v, false, b0.v,
                                                           (short)0, acc10, false, false);
        }
        if (n1) {
            acc01 = __builtin_amdgcn_wmma_f32_16x16x32_f16(false, a0.v, false, b1.v,
                                                           (short)0, acc01, false, false);
            acc11 = __builtin_amdgcn_wmma_f32_16x16x32_f16(false, a1.v, false, b1.v,
                                                           (short)0, acc11, false, false);
        }
        __syncthreads();                            // protect buffer overwrite
    }

    const int cm = hi * 8;                          // C row sub-offset (element r -> cm+r)
    const int cn = lane & 15;                       // C column
    auto emit = [&](f8v& acc, int mb, int nb) {
        int n = nb + cn;
        float bs = bias ? bias[n] : 0.f;
        #pragma unroll
        for (int r = 0; r < 8; ++r) {
            float v = acc[r] + bs;
            if (act == 1)      v = 0.5f * v * (1.f + erff(v * 0.7071067811865475f)); // exact GELU
            else if (act == 2) v = tanhf(v);
            else if (act == 3) v = fmaxf(v, 0.f);
            C[(long)(mb + cm + r) * ldc + n] = v;
        }
    };
    if (n0) {
        emit(acc00, mBlock + mBase,      nBlock + nBase);
        emit(acc10, mBlock + mBase + 16, nBlock + nBase);
    }
    if (n1) {
        emit(acc01, mBlock + mBase,      nBlock + nBase + 16);
        emit(acc11, mBlock + mBase + 16, nBlock + nBase + 16);
    }
}

// ---------------------------------------------------------------------------
// Elementwise / conversion kernels
// ---------------------------------------------------------------------------
__global__ void f32_to_f16_kernel(const float* __restrict__ s, _Float16* __restrict__ d, long n)
{
    long i = (long)blockIdx.x * blockDim.x + threadIdx.x;
    if (i < n) d[i] = (_Float16)s[i];
}

// W [K,N] f32 -> WT [N,K] f16
__global__ void transpose_w_kernel(const float* __restrict__ W, _Float16* __restrict__ WT,
                                   int K, int N)
{
    long i = (long)blockIdx.x * blockDim.x + threadIdx.x;
    if (i >= (long)K * N) return;
    int k = (int)(i / N);
    int n = (int)(i - (long)k * N);
    WT[(long)n * K + k] = (_Float16)W[i];
}

__global__ void fill_kernel(float* __restrict__ p, float v, long n)
{
    long i = (long)blockIdx.x * blockDim.x + threadIdx.x;
    if (i < n) p[i] = v;
}

// h[n,:] += concat(temporal_pe[time_ids[n]], spatial_pe[node_ids[n]])
__global__ __launch_bounds__(256) void add_pe_kernel(float* __restrict__ h,
                                                     const float* __restrict__ tpe,
                                                     const float* __restrict__ spe,
                                                     const int* __restrict__ tid,
                                                     const int* __restrict__ nid)
{
    int n = blockIdx.x, d = threadIdx.x;
    float pe = (d < 128) ? tpe[(long)tid[n] * 128 + d]
                         : spe[(long)nid[n] * 128 + (d - 128)];
    h[(long)n * 256 + d] += pe;
}

// LayerNorm over D (<=256) of x (+res) (+bvec), out = LN(..)*g+b, optional relu
__global__ __launch_bounds__(256) void ln_kernel(const float* __restrict__ x,
                                                 const float* __restrict__ res,
                                                 const float* __restrict__ bvec,
                                                 const float* __restrict__ g,
                                                 const float* __restrict__ b,
                                                 float* __restrict__ outp, int D, int act)
{
    __shared__ float sh[256];
    int row = blockIdx.x, t = threadIdx.x;
    long idx = (long)row * D + t;
    float v = 0.f;
    if (t < D) {
        v = x[idx];
        if (res)  v += res[idx];
        if (bvec) v += bvec[t];
    }
    sh[t] = (t < D) ? v : 0.f;
    __syncthreads();
    for (int o = 128; o > 0; o >>= 1) { if (t < o) sh[t] += sh[t + o]; __syncthreads(); }
    float mean = sh[0] / D;
    __syncthreads();
    float dv = (t < D) ? (v - mean) : 0.f;
    sh[t] = dv * dv;
    __syncthreads();
    for (int o = 128; o > 0; o >>= 1) { if (t < o) sh[t] += sh[t + o]; __syncthreads(); }
    float var = sh[0] / D;
    if (t < D) {
        float r = (v - mean) * rsqrtf(var + 1e-5f) * g[t] + b[t];
        if (act == 3) r = fmaxf(r, 0.f);
        outp[idx] = r;
    }
}

// ---------------------------------------------------------------------------
// GAT kernels
// ---------------------------------------------------------------------------
// a_s[n,h] = sum_d xl[n,h,d]*asrc[h,d] ; a_d likewise. One wave per head.
__global__ __launch_bounds__(256) void gat_scores_kernel(const float* __restrict__ xl,
                                                         const float* __restrict__ asrc,
                                                         const float* __restrict__ adst,
                                                         float* __restrict__ a_s,
                                                         float* __restrict__ a_d)
{
    int n = blockIdx.x;
    int w = threadIdx.x >> 5;   // head
    int l = threadIdx.x & 31;
    const float* xr = xl + (long)n * 2048 + w * 256;
    float s = 0.f, dd = 0.f;
    for (int i = l; i < 256; i += 32) {
        float v = xr[i];
        s  += v * asrc[w * 256 + i];
        dd += v * adst[w * 256 + i];
    }
    for (int o = 16; o > 0; o >>= 1) {
        s  += __shfl_down(s, o, 32);
        dd += __shfl_down(dd, o, 32);
    }
    if (l == 0) { a_s[n * 8 + w] = s; a_d[n * 8 + w] = dd; }
}

__device__ inline void atomicMaxF(float* addr, float v)
{
    if (v >= 0.f) atomicMax((int*)addr, __float_as_int(v));
    else          atomicMin((unsigned int*)addr, __float_as_uint(v));
}

// pass 1: e = leaky_relu(a_s[src]+a_d[dst]); segment max into m[dst]
__global__ void edge_pass1_kernel(const int* __restrict__ ei,
                                  const float* __restrict__ a_s, const float* __restrict__ a_d,
                                  float* __restrict__ ebuf, float* __restrict__ m,
                                  int E, int Nn)
{
    long t = (long)blockIdx.x * blockDim.x + threadIdx.x;
    if (t >= (long)(E + Nn) * 8) return;
    int e = (int)(t >> 3), hh = (int)(t & 7);
    int src = (e < E) ? ei[e]     : (e - E);
    int dst = (e < E) ? ei[E + e] : (e - E);
    float v = a_s[src * 8 + hh] + a_d[dst * 8 + hh];
    if (v < 0.f) v *= 0.2f;
    ebuf[(long)e * 8 + hh] = v;
    atomicMaxF(&m[dst * 8 + hh], v);
}

// pass 2: p = exp(e - m[dst]); z[dst] += p
__global__ void edge_pass2_kernel(const int* __restrict__ ei, float* __restrict__ ebuf,
                                  const float* __restrict__ m, float* __restrict__ z,
                                  int E, int Nn)
{
    long t = (long)blockIdx.x * blockDim.x + threadIdx.x;
    if (t >= (long)(E + Nn) * 8) return;
    int e = (int)(t >> 3), hh = (int)(t & 7);
    int dst = (e < E) ? ei[E + e] : (e - E);
    float p = expf(ebuf[(long)e * 8 + hh] - m[dst * 8 + hh]);
    ebuf[(long)e * 8 + hh] = p;
    atomicAdd(&z[dst * 8 + hh], p);
}

// agg[dst,d] += (1/H) * sum_h alpha[e,h] * xl[src,h,d]
__global__ __launch_bounds__(256) void gat_agg_kernel(const int* __restrict__ ei,
                                                      const float* __restrict__ xl,
                                                      const float* __restrict__ p,
                                                      const float* __restrict__ z,
                                                      float* __restrict__ agg, int E, int Nn)
{
    int e = blockIdx.x, d = threadIdx.x;
    int src = (e < E) ? ei[e]     : (e - E);
    int dst = (e < E) ? ei[E + e] : (e - E);
    float s = 0.f;
    #pragma unroll
    for (int hh = 0; hh < 8; ++hh) {
        float alpha = p[(long)e * 8 + hh] / (z[dst * 8 + hh] + 1e-16f);
        s += alpha * xl[(long)src * 2048 + hh * 256 + d];
    }
    atomicAdd(&agg[(long)dst * 256 + d], s * 0.125f);
}

// ---------------------------------------------------------------------------
// Attention helpers
// ---------------------------------------------------------------------------
// v [N,256] f32 -> v16t [(b,h,d), nloc] f16  (V^T per head, row length NPG)
__global__ void v_transpose_kernel(const float* __restrict__ v, _Float16* __restrict__ vt)
{
    long i = (long)blockIdx.x * blockDim.x + threadIdx.x;
    if (i >= (long)4096 * 256) return;
    int node = (int)(i >> 8);
    int c    = (int)(i & 255);
    int hh = c >> 5, dd = c & 31;
    int b = node >> 9, nl = node & 511;
    vt[((long)(b * 8 + hh) * 32 + dd) * 512 + nl] = (_Float16)v[i];
}

// row softmax over 512 cols with scale; writes f16 probabilities
__global__ __launch_bounds__(256) void attn_softmax_kernel(float* __restrict__ sc,
                                                           _Float16* __restrict__ p16,
                                                           float scale)
{
    __shared__ float sh[256];
    long row = blockIdx.x;
    float* r = sc + row * 512;
    int t = threadIdx.x;
    float v0 = r[t] * scale, v1 = r[t + 256] * scale;
    sh[t] = fmaxf(v0, v1);
    __syncthreads();
    for (int o = 128; o > 0; o >>= 1) { if (t < o) sh[t] = fmaxf(sh[t], sh[t + o]); __syncthreads(); }
    float mx = sh[0];
    __syncthreads();
    float e0 = expf(v0 - mx), e1 = expf(v1 - mx);
    sh[t] = e0 + e1;
    __syncthreads();
    for (int o = 128; o > 0; o >>= 1) { if (t < o) sh[t] += sh[t + o]; __syncthreads(); }
    float inv = 1.f / sh[0];
    p16[row * 512 + t]       = (_Float16)(e0 * inv);
    p16[row * 512 + t + 256] = (_Float16)(e1 * inv);
}

// ---------------------------------------------------------------------------
// Pooling / classifier
// ---------------------------------------------------------------------------
__global__ __launch_bounds__(256) void pool_softmax_kernel(const float* __restrict__ s,
                                                           float* __restrict__ w)
{
    __shared__ float sh[256];
    int b = blockIdx.x, t = threadIdx.x;
    float v0 = s[b * 512 + t], v1 = s[b * 512 + t + 256];
    sh[t] = fmaxf(v0, v1);
    __syncthreads();
    for (int o = 128; o > 0; o >>= 1) { if (t < o) sh[t] = fmaxf(sh[t], sh[t + o]); __syncthreads(); }
    float mx = sh[0];
    __syncthreads();
    float e0 = expf(v0 - mx), e1 = expf(v1 - mx);
    sh[t] = e0 + e1;
    __syncthreads();
    for (int o = 128; o > 0; o >>= 1) { if (t < o) sh[t] += sh[t + o]; __syncthreads(); }
    float inv = 1.f / sh[0];
    w[b * 512 + t]       = e0 * inv;
    w[b * 512 + t + 256] = e1 * inv;
}

__global__ __launch_bounds__(256) void pool_emb_kernel(const float* __restrict__ w,
                                                       const float* __restrict__ h,
                                                       float* __restrict__ emb)
{
    int b = blockIdx.x, d = threadIdx.x;
    const float* hb = h + (long)b * 512 * 256;
    const float* wb = w + b * 512;
    float acc = 0.f;
    for (int n = 0; n < 512; ++n) acc += wb[n] * hb[(long)n * 256 + d];
    emb[b * 256 + d] = acc;
}

// tiny GEMM: out[m,j] = act(sum_k A[m,k]*W[k,j] + bias[j]), one block per row
__global__ void small_mm_kernel(const float* __restrict__ A, const float* __restrict__ W,
                                const float* __restrict__ bias, float* __restrict__ outp,
                                int K, int Nn, int act)
{
    int m = blockIdx.x, j = threadIdx.x;
    if (j >= Nn) return;
    float s = bias ? bias[j] : 0.f;
    for (int k = 0; k < K; ++k) s += A[(long)m * K + k] * W[(long)k * Nn + j];
    if (act == 2)      s = tanhf(s);
    else if (act == 3) s = fmaxf(s, 0.f);
    outp[(long)m * Nn + j] = s;
}

// ---------------------------------------------------------------------------
// Host orchestration
// ---------------------------------------------------------------------------
extern "C" void kernel_launch(void* const* d_in, const int* in_sizes, int n_in,
                              void* d_out, int out_size, void* d_ws, size_t ws_size,
                              hipStream_t stream)
{
    (void)in_sizes; (void)n_in; (void)out_size; (void)ws_size;
    const int N = 4096, E = 32768, B = 8, NPG = 512, D = 256, H = 8, DH = 32,
              L = 6, FF = 1024, NF = 64;
    const int EC = E + N;

    const float* x        = (const float*)d_in[0];
    const int*   ei       = (const int*)  d_in[1];
    const int*   node_ids = (const int*)  d_in[3];
    const int*   time_ids = (const int*)  d_in[4];
    const float* w_in     = (const float*)d_in[5];
    const float* b_in     = (const float*)d_in[6];
    const float* lnin_g   = (const float*)d_in[7];
    const float* lnin_b   = (const float*)d_in[8];
    const float* tpe      = (const float*)d_in[9];
    const float* spe      = (const float*)d_in[10];
    const float* gat_W    = (const float*)d_in[11];
    const float* gat_asrc = (const float*)d_in[12];
    const float* gat_adst = (const float*)d_in[13];
    const float* gat_b    = (const float*)d_in[14];
    const float* wq = (const float*)d_in[15], *wk = (const float*)d_in[16];
    const float* wv = (const float*)d_in[17], *wo = (const float*)d_in[18];
    const float* bq = (const float*)d_in[19], *bk = (const float*)d_in[20];
    const float* bv = (const float*)d_in[21], *bo = (const float*)d_in[22];
    const float* ln1g = (const float*)d_in[23], *ln2g = (const float*)d_in[24], *ln3g = (const float*)d_in[25];
    const float* ln1b = (const float*)d_in[26], *ln2b = (const float*)d_in[27], *ln3b = (const float*)d_in[28];
    const float* w1 = (const float*)d_in[29], *b1 = (const float*)d_in[30];
    const float* w2 = (const float*)d_in[31], *b2 = (const float*)d_in[32];
    const float* wp1 = (const float*)d_in[33], *bp1 = (const float*)d_in[34];
    const float* wp2 = (const float*)d_in[35], *bp2 = (const float*)d_in[36];
    const float* wc1 = (const float*)d_in[37], *bc1 = (const float*)d_in[38];
    const float* lnc1g = (const float*)d_in[39], *lnc1b = (const float*)d_in[40];
    const float* wc2 = (const float*)d_in[41], *bc2 = (const float*)d_in[42];
    const float* lnc2g = (const float*)d_in[43], *lnc2b = (const float*)d_in[44];
    const float* wc3 = (const float*)d_in[45], *bc3 = (const float*)d_in[46];
    float* out = (float*)d_out;

    // workspace carve-up (256B aligned slots)
    char* wsb = (char*)d_ws;
    size_t off = 0;
    auto carve = [&](size_t bytes) -> void* {
        void* p = wsb + off;
        off += (bytes + 255) & ~(size_t)255;
        return p;
    };
    float*    h      = (float*)   carve((size_t)N * D * 4);
    _Float16* hf16   = (_Float16*)carve((size_t)N * FF * 2);      // f16 A-operand buffer
    _Float16* wt16   = (_Float16*)carve((size_t)2048 * 256 * 2);  // transposed-weight buffer
    float*    xl     = (float*)   carve((size_t)N * H * D * 4);
    float*    a_s    = (float*)   carve((size_t)N * H * 4);
    float*    a_d    = (float*)   carve((size_t)N * H * 4);
    float*    mbuf   = (float*)   carve((size_t)N * H * 4);
    float*    zbuf   = (float*)   carve((size_t)N * H * 4);
    float*    ebuf   = (float*)   carve((size_t)EC * H * 4);
    float*    agg    = (float*)   carve((size_t)N * D * 4);
    float*    qf     = (float*)   carve((size_t)N * D * 4);
    float*    kf     = (float*)   carve((size_t)N * D * 4);
    float*    vf     = (float*)   carve((size_t)N * D * 4);
    _Float16* q16    = (_Float16*)carve((size_t)N * D * 2);
    _Float16* k16    = (_Float16*)carve((size_t)N * D * 2);
    _Float16* v16t   = (_Float16*)carve((size_t)N * D * 2);
    float*    scores = (float*)   carve((size_t)B * H * NPG * NPG * 4);
    _Float16* p16    = (_Float16*)carve((size_t)B * H * NPG * NPG * 2);
    float*    obuf   = (float*)   carve((size_t)N * D * 4);
    float*    fbuf   = (float*)   carve((size_t)N * FF * 4);
    float*    hp     = (float*)   carve((size_t)N * 128 * 4);
    float*    sbuf   = (float*)   carve((size_t)N * 4);
    float*    wbuf   = (float*)   carve((size_t)N * 4);
    float*    emb    = (float*)   carve((size_t)B * D * 4);
    float*    c1     = (float*)   carve((size_t)B * 128 * 4);
    float*    c2     = (float*)   carve((size_t)B * 64 * 4);

    auto gemm = [&](const _Float16* A, int lda, long sAo, long sAi,
                    const _Float16* BT, int ldb, long sBo, long sBi,
                    float* C, int ldc, long sCo, long sCi,
                    const float* bias, int M, int Nn, int K, int act, int Z, int innerN) {
        dim3 g((M + 127) / 128, (Nn + 63) / 64, Z);
        gemm_wmma_kernel<<<g, 256, 0, stream>>>(A, lda, sAo, sAi, BT, ldb, sBo, sBi,
                                                C, ldc, sCo, sCi, bias, M, Nn, K, act, innerN);
    };
    auto cvt = [&](const float* s, _Float16* d, long n) {
        f32_to_f16_kernel<<<(int)((n + 255) / 256), 256, 0, stream>>>(s, d, n);
    };
    auto trans = [&](const float* W, int K, int Nn) {
        transpose_w_kernel<<<(int)(((long)K * Nn + 255) / 256), 256, 0, stream>>>(W, wt16, K, Nn);
    };
    auto fill = [&](float* p, float v, long n) {
        fill_kernel<<<(int)((n + 255) / 256), 256, 0, stream>>>(p, v, n);
    };

    // ---- input projection + LN + positional encodings ----
    cvt(x, hf16, (long)N * NF);
    trans(w_in, NF, D);
    gemm(hf16, NF, 0, 0, wt16, NF, 0, 0, h, D, 0, 0, b_in, N, D, NF, 0, 1, 1);
    ln_kernel<<<N, 256, 0, stream>>>(h, nullptr, nullptr, lnin_g, lnin_b, h, D, 0);
    add_pe_kernel<<<N, 256, 0, stream>>>(h, tpe, spe, time_ids, node_ids);

    const int tE = EC * H;
    for (int l = 0; l < L; ++l) {
        // ---------------- GATConv ----------------
        cvt(h, hf16, (long)N * D);
        trans(gat_W + (size_t)l * D * H * D, D, H * D);
        gemm(hf16, D, 0, 0, wt16, D, 0, 0, xl, H * D, 0, 0, nullptr, N, H * D, D, 0, 1, 1);
        gat_scores_kernel<<<N, 256, 0, stream>>>(xl, gat_asrc + (size_t)l * H * D,
                                                 gat_adst + (size_t)l * H * D, a_s, a_d);
        fill(mbuf, -3.0e38f, (long)N * H);
        fill(zbuf, 0.f, (long)N * H);
        edge_pass1_kernel<<<(tE + 255) / 256, 256, 0, stream>>>(ei, a_s, a_d, ebuf, mbuf, E, N);
        edge_pass2_kernel<<<(tE + 255) / 256, 256, 0, stream>>>(ei, ebuf, mbuf, zbuf, E, N);
        fill(agg, 0.f, (long)N * D);
        gat_agg_kernel<<<EC, 256, 0, stream>>>(ei, xl, ebuf, zbuf, agg, E, N);
        ln_kernel<<<N, 256, 0, stream>>>(h, agg, gat_b + (size_t)l * D,
                                         ln1g + (size_t)l * D, ln1b + (size_t)l * D, h, D, 0);

        // ---------------- per-graph self-attention ----------------
        cvt(h, hf16, (long)N * D);
        trans(wq + (size_t)l * D * D, D, D);
        gemm(hf16, D, 0, 0, wt16, D, 0, 0, qf, D, 0, 0, bq + (size_t)l * D, N, D, D, 0, 1, 1);
        trans(wk + (size_t)l * D * D, D, D);
        gemm(hf16, D, 0, 0, wt16, D, 0, 0, kf, D, 0, 0, bk + (size_t)l * D, N, D, D, 0, 1, 1);
        trans(wv + (size_t)l * D * D, D, D);
        gemm(hf16, D, 0, 0, wt16, D, 0, 0, vf, D, 0, 0, bv + (size_t)l * D, N, D, D, 0, 1, 1);
        cvt(qf, q16, (long)N * D);
        cvt(kf, k16, (long)N * D);
        v_transpose_kernel<<<((long)N * D + 255) / 256, 256, 0, stream>>>(vf, v16t);

        long qs_o = (long)NPG * D, qs_i = DH;                 // head offsets into [N,D]
        long sc_i = (long)NPG * NPG, sc_o = sc_i * H;         // scores/probs offsets
        long vt_i = (long)DH * NPG, vt_o = vt_i * H;          // V^T offsets
        // scores[b,h] = Q_bh (512x32) @ K_bh^T
        gemm(q16, D, qs_o, qs_i, k16, D, qs_o, qs_i,
             scores, NPG, sc_o, sc_i, nullptr, NPG, NPG, DH, 0, B * H, H);
        attn_softmax_kernel<<<B * H * NPG, 256, 0, stream>>>(scores, p16, 0.17677669529663687f);
        // o[b,h] = P (512x512) @ V_bh (512x32)
        gemm(p16, NPG, sc_o, sc_i, v16t, NPG, vt_o, vt_i,
             obuf, D, qs_o, qs_i, nullptr, NPG, DH, NPG, 0, B * H, H);
        // output projection
        cvt(obuf, hf16, (long)N * D);
        trans(wo + (size_t)l * D * D, D, D);
        gemm(hf16, D, 0, 0, wt16, D, 0, 0, qf, D, 0, 0, bo + (size_t)l * D, N, D, D, 0, 1, 1);
        ln_kernel<<<N, 256, 0, stream>>>(h, qf, nullptr,
                                         ln2g + (size_t)l * D, ln2b + (size_t)l * D, h, D, 0);

        // ---------------- FFN ----------------
        cvt(h, hf16, (long)N * D);
        trans(w1 + (size_t)l * D * FF, D, FF);
        gemm(hf16, D, 0, 0, wt16, D, 0, 0, fbuf, FF, 0, 0, b1 + (size_t)l * FF, N, FF, D, 1, 1, 1);
        cvt(fbuf, hf16, (long)N * FF);
        trans(w2 + (size_t)l * FF * D, FF, D);
        gemm(hf16, FF, 0, 0, wt16, FF, 0, 0, obuf, D, 0, 0, b2 + (size_t)l * D, N, D, FF, 0, 1, 1);
        ln_kernel<<<N, 256, 0, stream>>>(h, obuf, nullptr,
                                         ln3g + (size_t)l * D, ln3b + (size_t)l * D, h, D, 0);
    }

    // ---- attention pooling ----
    cvt(h, hf16, (long)N * D);
    trans(wp1, D, 128);
    gemm(hf16, D, 0, 0, wt16, D, 0, 0, hp, 128, 0, 0, bp1, N, 128, D, 2, 1, 1); // tanh
    small_mm_kernel<<<N, 32, 0, stream>>>(hp, wp2, bp2, sbuf, 128, 1, 0);
    pool_softmax_kernel<<<B, 256, 0, stream>>>(sbuf, wbuf);
    pool_emb_kernel<<<B, 256, 0, stream>>>(wbuf, h, emb);

    // ---- classifier ----
    small_mm_kernel<<<B, 128, 0, stream>>>(emb, wc1, bc1, c1, D, 128, 0);
    ln_kernel<<<B, 256, 0, stream>>>(c1, nullptr, nullptr, lnc1g, lnc1b, c1, 128, 3);
    small_mm_kernel<<<B, 64, 0, stream>>>(c1, wc2, bc2, c2, 128, 64, 0);
    ln_kernel<<<B, 256, 0, stream>>>(c2, nullptr, nullptr, lnc2g, lnc2b, c2, 64, 3);
    small_mm_kernel<<<B, 32, 0, stream>>>(c2, wc3, bc3, out, 64, 10, 0);
}
